// MultiLatentAttention_39333310496902
// MI455X (gfx1250) — matrix operand order repaired and checked by reading, not compiled, not run on hardware
//
#include <hip/hip_runtime.h>

#define B_   2
#define S_   2048
#define D_   2048
#define H_   16
#define HKV_ 4
#define RQ_  1024
#define RKV_ 512
#define DH_  192
#define DR_  64
#define DN_  128
#define DV_  128

typedef __bf16 bf16;
typedef __attribute__((ext_vector_type(16))) __bf16 v16bf;
typedef __attribute__((ext_vector_type(8)))  __bf16 v8bf;
typedef __attribute__((ext_vector_type(8)))  float  v8f;
typedef __attribute__((ext_vector_type(4)))  int    v4i;

#if __has_builtin(__builtin_amdgcn_global_load_async_to_lds_b128)
#define HAS_ASYNC_LDS 1
#else
#define HAS_ASYNC_LDS 0
#endif

__device__ __forceinline__ bf16 f2bf(float f) {
  unsigned int u = __builtin_bit_cast(unsigned int, f);
  u += 0x7FFFu + ((u >> 16) & 1u);              // round-to-nearest-even
  unsigned short s = (unsigned short)(u >> 16);
  return __builtin_bit_cast(bf16, s);
}
__device__ __forceinline__ float bf2f(bf16 b) {
  unsigned int u = ((unsigned int)__builtin_bit_cast(unsigned short, b)) << 16;
  return __builtin_bit_cast(float, u);
}
// Two 16-byte-aligned 8-element loads -> one WMMA operand (emits b128 loads).
__device__ __forceinline__ v16bf ld_frag16(const bf16* p0, const bf16* p1) {
  v8bf a = *reinterpret_cast<const v8bf*>(p0);
  v8bf b = *reinterpret_cast<const v8bf*>(p1);
  return __builtin_shufflevector(a, b, 0, 1, 2, 3, 4, 5, 6, 7,
                                 8, 9, 10, 11, 12, 13, 14, 15);
}

// 16B global -> LDS copy: async (GLOBAL_LOAD_ASYNC_TO_LDS_B128) if available.
__device__ __forceinline__ void async_copy_b128(const bf16* gsrc, bf16* ldst) {
#if HAS_ASYNC_LDS
  typedef __attribute__((address_space(1))) v4i* gp_t;
  typedef __attribute__((address_space(3))) v4i* lp_t;
  __builtin_amdgcn_global_load_async_to_lds_b128((gp_t)gsrc, (lp_t)ldst, 0, 0);
#else
  *reinterpret_cast<uint4*>(ldst) = *reinterpret_cast<const uint4*>(gsrc);
#endif
}
__device__ __forceinline__ void async_wait() {
#if HAS_ASYNC_LDS
#if __has_builtin(__builtin_amdgcn_s_wait_asynccnt)
  __builtin_amdgcn_s_wait_asynccnt(0);
#else
  asm volatile("s_wait_asynccnt 0" ::: "memory");
#endif
#endif
}

// ---------------------------------------------------------------- cvt f32->bf16 (8 elems/iter)
__global__ void cvt_f32_bf16(const float* __restrict__ in, bf16* __restrict__ out, size_t n8) {
  for (size_t i = (size_t)blockIdx.x * blockDim.x + threadIdx.x; i < n8;
       i += (size_t)gridDim.x * blockDim.x) {
    const float4* s = reinterpret_cast<const float4*>(in) + 2 * i;
    float4 a = s[0], b = s[1];
    v8bf r;
    r[0] = f2bf(a.x); r[1] = f2bf(a.y); r[2] = f2bf(a.z); r[3] = f2bf(a.w);
    r[4] = f2bf(b.x); r[5] = f2bf(b.y); r[6] = f2bf(b.z); r[7] = f2bf(b.w);
    reinterpret_cast<v8bf*>(out)[i] = r;
  }
}

// ---------------------------------------------------------------- RMSNorm (one block per row)
__global__ __launch_bounds__(256) void rmsnorm_bf16(const bf16* __restrict__ in, int in_ld,
                                                    const float* __restrict__ g,
                                                    bf16* __restrict__ out, int cols) {
  const int row = blockIdx.x;
  const bf16* x = in + (size_t)row * in_ld;
  const int cols8 = cols >> 3;
  float ss = 0.f;
  for (int c8 = threadIdx.x; c8 < cols8; c8 += 256) {
    v8bf xv = reinterpret_cast<const v8bf*>(x)[c8];
#pragma unroll
    for (int j = 0; j < 8; ++j) { float v = bf2f(xv[j]); ss += v * v; }
  }
  __shared__ float red[256];
  red[threadIdx.x] = ss;
  __syncthreads();
  for (int s = 128; s > 0; s >>= 1) {
    if (threadIdx.x < s) red[threadIdx.x] += red[threadIdx.x + s];
    __syncthreads();
  }
  const float scale = rsqrtf(red[0] / (float)cols + 1e-20f);
  for (int c8 = threadIdx.x; c8 < cols8; c8 += 256) {
    v8bf xv = reinterpret_cast<const v8bf*>(x)[c8];
    const float4* gp = reinterpret_cast<const float4*>(g) + 2 * c8;
    float4 g0 = gp[0], g1 = gp[1];
    v8bf r;
    r[0] = f2bf(bf2f(xv[0]) * scale * g0.x);
    r[1] = f2bf(bf2f(xv[1]) * scale * g0.y);
    r[2] = f2bf(bf2f(xv[2]) * scale * g0.z);
    r[3] = f2bf(bf2f(xv[3]) * scale * g0.w);
    r[4] = f2bf(bf2f(xv[4]) * scale * g1.x);
    r[5] = f2bf(bf2f(xv[5]) * scale * g1.y);
    r[6] = f2bf(bf2f(xv[6]) * scale * g1.z);
    r[7] = f2bf(bf2f(xv[7]) * scale * g1.w);
    reinterpret_cast<v8bf*>(out + (size_t)row * cols)[c8] = r;
  }
}

// ---------------------------------------------------------------- generic bf16 WMMA GEMM
// C[M,N] = A[M,K] @ B[K,N] + bias[N]; A,B bf16 row-major; out bf16 or f32.
// Block 256 thr = 8 waves; tile 128x128; wave w -> rows [16w,16w+16), all 128 cols.
// Double-buffered LDS, A tile via async-to-LDS, B tile register-pipelined so the
// WMMA chain's dscnt waits only cover its own fragment loads.
__global__ __launch_bounds__(256) void gemm_bf16(const bf16* __restrict__ A,
                                                 const bf16* __restrict__ Bm,
                                                 const float* __restrict__ bias,
                                                 void* __restrict__ Cout,
                                                 int M, int N, int K, int outF32) {
  __shared__ __align__(16) bf16 As[2][128 * 40];   // [m][k] 32 + pad 8
  __shared__ __align__(16) bf16 Bt[2][128 * 40];   // [n][k] 32 + pad 8 (transposed)

  const int m0   = blockIdx.y * 128;
  const int n0   = blockIdx.x * 128;
  const int tid  = threadIdx.x;
  const int lane = tid & 31;
  const int wv   = tid >> 5;      // 0..7
  const int half = lane >> 4;     // 0/1
  const int lx   = lane & 15;

  const int arow = tid >> 1;            // A: 0..127
  const int acg  = (tid & 1) << 4;
  const int brow = tid >> 3;            // B: k row 0..31
  const int bcg  = (tid & 7) << 4;      // B: n chunk
  const bool bok = (n0 + bcg + 16 <= N);

  auto issue_A = [&](int k0, int p) {   // async 2x16B per thread
    const bf16* g = A + (size_t)(m0 + arow) * K + k0 + acg;
    bf16* l = &As[p][arow * 40 + acg];
    async_copy_b128(g, l);
    async_copy_b128(g + 8, l + 8);
  };
  auto load_B = [&](int k0, v8bf& x0, v8bf& x1) {
#pragma unroll
    for (int j = 0; j < 8; ++j) { x0[j] = __builtin_bit_cast(bf16, (unsigned short)0);
                                  x1[j] = __builtin_bit_cast(bf16, (unsigned short)0); }
    if (bok) {
      const v8bf* src = reinterpret_cast<const v8bf*>(Bm + (size_t)(k0 + brow) * N + n0 + bcg);
      x0 = src[0];
      x1 = src[1];
    }
  };
  auto scatter_B = [&](int p, v8bf x0, v8bf x1) {
#pragma unroll
    for (int j = 0; j < 8; ++j) Bt[p][(bcg + j) * 40 + brow]     = x0[j];
#pragma unroll
    for (int j = 0; j < 8; ++j) Bt[p][(bcg + 8 + j) * 40 + brow] = x1[j];
  };

  v8f acc[8];
#pragma unroll
  for (int i = 0; i < 8; ++i)
#pragma unroll
    for (int j = 0; j < 8; ++j) acc[i][j] = 0.f;

  {   // prologue: stage tile 0
    issue_A(0, 0);
    v8bf x0, x1;
    load_B(0, x0, x1);
    scatter_B(0, x0, x1);
    async_wait();
  }
  __syncthreads();

  const int nsteps = K >> 5;
  for (int st = 0; st < nsteps; ++st) {
    const int p = st & 1;
    const bool more = (st + 1 < nsteps);
    v8bf nx0, nx1;
    if (more) {                         // issue next tile's global traffic early
      const int kn = (st + 1) << 5;
      issue_A(kn, p ^ 1);
      load_B(kn, nx0, nx1);
      if (st + 2 < nsteps) {
        __builtin_prefetch(A + (size_t)(m0 + arow) * K + ((st + 2) << 5) + acg, 0, 1);
        if (bok)
          __builtin_prefetch(Bm + (size_t)(((st + 2) << 5) + brow) * N + n0 + bcg, 0, 1);
      }
    }

    // A fragment (16x32): lanes 0-15: K 0-15; lanes 16-31: K 16-31
    const bf16* abase = &As[p][(wv * 16 + lx) * 40];
    v16bf af = ld_frag16(abase + half * 16, abase + 8 + half * 16);

    v16bf bfr[8];
#pragma unroll
    for (int nt = 0; nt < 8; ++nt) {
      const bf16* bbase = &Bt[p][(nt * 16 + lx) * 40 + half * 16];
      bfr[nt] = ld_frag16(bbase, bbase + 8);
    }
#pragma unroll
    for (int nt = 0; nt < 8; ++nt)
      acc[nt] = __builtin_amdgcn_wmma_f32_16x16x32_bf16(false, af, false, bfr[nt],
                                                        (short)0, acc[nt], false, false);

    if (more) scatter_B(p ^ 1, nx0, nx1);   // LDS stores after the WMMA chain
    async_wait();
    __syncthreads();
  }

#pragma unroll
  for (int nt = 0; nt < 8; ++nt) {
    if (n0 + nt * 16 >= N) continue;
    const int gc = n0 + nt * 16 + lx;
    const float bv = bias[gc];
#pragma unroll
    for (int v = 0; v < 8; ++v) {
      const int gr = m0 + wv * 16 + v + half * 8;   // C layout: VGPR v -> M = v + 8*half
      const float val = acc[nt][v] + bv;
      if (outF32) reinterpret_cast<float*>(Cout)[(size_t)gr * N + gc] = val;
      else        reinterpret_cast<bf16*>(Cout)[(size_t)gr * N + gc]  = f2bf(val);
    }
  }
}

// ---------------------------------------------------------------- RoPE + assemble Q
// qb: [B*S, H*192] per-head [nrope(128) | rope_in(64)] -> Q: [B,H,S,192] = [roped(64)|nrope(128)]
__global__ void assemble_q_k(const bf16* __restrict__ qb, bf16* __restrict__ Q) {
  const size_t total = (size_t)B_ * H_ * S_ * DH_;
  for (size_t idx = (size_t)blockIdx.x * blockDim.x + threadIdx.x; idx < total;
       idx += (size_t)gridDim.x * blockDim.x) {
    int d = (int)(idx % DH_);
    size_t t = idx / DH_;
    int s = (int)(t % S_); t /= S_;
    int h = (int)(t % H_);
    int b = (int)(t / H_);
    const bf16* src = qb + (size_t)(b * S_ + s) * (H_ * DH_) + h * DH_;
    float val;
    if (d < DR_) {
      int i = d & 31;
      float inv = __expf(-(float)i * (9.21034037f / 32.f));   // 10000^(-i/32)
      float ang = (float)s * inv;
      float sn, cs;
      __sincosf(ang, &sn, &cs);
      float x1 = bf2f(src[DN_ + d]);
      if (d < 32) { float x2 = bf2f(src[DN_ + d + 32]); val = x1 * cs - x2 * sn; }
      else        { float x2 = bf2f(src[DN_ + d - 32]); val = x1 * cs + x2 * sn; }
    } else {
      val = bf2f(src[d - DR_]);
    }
    Q[idx] = f2bf(val);
  }
}

// ---------------------------------------------------------------- assemble K (rope bcast + nrope) and V
__global__ void assemble_kv_k(const bf16* __restrict__ kvb, const bf16* __restrict__ kva,
                              bf16* __restrict__ Kt, bf16* __restrict__ Vt) {
  const int per = DH_ + DV_;   // 320
  const size_t total = (size_t)B_ * HKV_ * S_ * per;
  for (size_t idx = (size_t)blockIdx.x * blockDim.x + threadIdx.x; idx < total;
       idx += (size_t)gridDim.x * blockDim.x) {
    int d = (int)(idx % per);
    size_t t = idx / per;
    int s = (int)(t % S_); t /= S_;
    int kvh = (int)(t % HKV_);
    int b = (int)(t / HKV_);
    const size_t kvbrow = (size_t)(b * S_ + s) * (HKV_ * (DN_ + DV_)) + kvh * (DN_ + DV_);
    if (d < DH_) {
      float val;
      if (d < DR_) {
        const bf16* kr = kva + (size_t)(b * S_ + s) * (RKV_ + DR_) + RKV_;
        int i = d & 31;
        float inv = __expf(-(float)i * (9.21034037f / 32.f));
        float ang = (float)s * inv;
        float sn, cs;
        __sincosf(ang, &sn, &cs);
        float x1 = bf2f(kr[d]);
        if (d < 32) { float x2 = bf2f(kr[d + 32]); val = x1 * cs - x2 * sn; }
        else        { float x2 = bf2f(kr[d - 32]); val = x1 * cs + x2 * sn; }
      } else {
        val = bf2f(kvb[kvbrow + (d - DR_)]);
      }
      Kt[((size_t)(b * HKV_ + kvh) * S_ + s) * DH_ + d] = f2bf(val);
    } else {
      int dv = d - DH_;
      Vt[((size_t)(b * HKV_ + kvh) * S_ + s) * DV_ + dv] = kvb[kvbrow + DN_ + dv];
    }
  }
}

// ---------------------------------------------------------------- flash attention
// Q [B,H,S,192], K [B,HKV,S,192], V [B,HKV,S,128] -> Out [B*S, H*128] bf16
// 128 thr = 4 waves; block = 64 query rows of one head; K tile via async-to-LDS,
// V tile transposed in LDS; fragments batched before their WMMA groups.
__global__ __launch_bounds__(128) void mla_attn(const bf16* __restrict__ Q,
                                                const bf16* __restrict__ Kt,
                                                const bf16* __restrict__ Vt,
                                                bf16* __restrict__ Out) {
  const int qt   = blockIdx.x;          // 0..S/64-1
  const int bh   = blockIdx.y;          // 0..B*H-1
  const int b    = bh >> 4;
  const int h    = bh & 15;
  const int kvh  = h >> 2;              // GROUPS = 4
  const int lane = threadIdx.x & 31;
  const int wv   = threadIdx.x >> 5;    // 0..3
  const int half = lane >> 4;
  const int lx   = lane & 15;

  __shared__ __align__(16) bf16 Ks[64 * 200];     // [key][d] 192 + pad 8
  __shared__ __align__(16) bf16 Vst[128 * 72];    // [dv][key] 64 + pad 8 (transposed)
  __shared__ __align__(16) bf16 Ps[4][16 * 72];   // per-wave P (16x64 + pad 8)

  // Q fragments: wave row = lx, 6 k-steps of 32
  v16bf qf[6];
  {
    const bf16* qrow = Q + ((size_t)(b * H_ + h) * S_ + qt * 64 + wv * 16 + lx) * DH_;
#pragma unroll
    for (int ks = 0; ks < 6; ++ks)
      qf[ks] = ld_frag16(qrow + 32 * ks + half * 16, qrow + 32 * ks + 8 + half * 16);
  }

  v8f o[8];
#pragma unroll
  for (int i = 0; i < 8; ++i)
#pragma unroll
    for (int j = 0; j < 8; ++j) o[i][j] = 0.f;
  float m_i[8], l_i[8];
#pragma unroll
  for (int v = 0; v < 8; ++v) { m_i[v] = -3.0e38f; l_i[v] = 0.f; }

  const bf16* Kbase = Kt + (size_t)(b * HKV_ + kvh) * S_ * DH_;
  const bf16* Vbase = Vt + (size_t)(b * HKV_ + kvh) * S_ * DV_;
  const float scl = 0.07216878364f;     // 1/sqrt(192)

  for (int kt = 0; kt <= qt; ++kt) {
    __syncthreads();
    for (int c = threadIdx.x; c < 64 * 24; c += 128) {      // K tile 64x192 (async)
      int row = c / 24, col = (c % 24) * 8;
      async_copy_b128(Kbase + (size_t)(kt * 64 + row) * DH_ + col, &Ks[row * 200 + col]);
    }
    for (int c = threadIdx.x; c < 64 * 16; c += 128) {      // V tile 64x128 -> transpose
      int key = c / 16, cg = (c % 16) * 8;
      v8bf x = *reinterpret_cast<const v8bf*>(Vbase + (size_t)(kt * 64 + key) * DV_ + cg);
#pragma unroll
      for (int j = 0; j < 8; ++j) Vst[(cg + j) * 72 + key] = x[j];
    }
    if (kt < qt) {      // prefetch next key tile into cache
      __builtin_prefetch(Kbase + (size_t)((kt + 1) * 64 + (threadIdx.x >> 1)) * DH_, 0, 1);
      __builtin_prefetch(Vbase + (size_t)((kt + 1) * 64 + (threadIdx.x >> 1)) * DV_, 0, 1);
    }
    async_wait();
    __syncthreads();

    // S = Q K^T : wave's 16 rows x 64 keys
    v8f sacc[4];
#pragma unroll
    for (int i = 0; i < 4; ++i)
#pragma unroll
      for (int j = 0; j < 8; ++j) sacc[i][j] = 0.f;
#pragma unroll
    for (int ks = 0; ks < 6; ++ks) {
      v16bf bk[4];
#pragma unroll
      for (int nt = 0; nt < 4; ++nt) {
        // B[k][n] = K[key = nt*16+lx][d = 32*ks + 16*half + e]  (contiguous in d)
        const bf16* kb = &Ks[(nt * 16 + lx) * 200 + 32 * ks + 16 * half];
        bk[nt] = ld_frag16(kb, kb + 8);
      }
#pragma unroll
      for (int nt = 0; nt < 4; ++nt)
        sacc[nt] = __builtin_amdgcn_wmma_f32_16x16x32_bf16(false, qf[ks], false, bk[nt],
                                                           (short)0, sacc[nt], false, false);
    }

    const bool diag = (kt == qt);
#pragma unroll
    for (int v = 0; v < 8; ++v) {
      const int grow = qt * 64 + wv * 16 + v + half * 8;
      float sv[4];
      float mx = -3.0e38f;
#pragma unroll
      for (int nt = 0; nt < 4; ++nt) {
        float x = sacc[nt][v] * scl;
        if (diag) {
          int gcol = kt * 64 + nt * 16 + lx;
          if (gcol > grow) x = -1.0e30f;
        }
        sv[nt] = x;
        mx = fmaxf(mx, x);
      }
#pragma unroll
      for (int ms = 8; ms >= 1; ms >>= 1) mx = fmaxf(mx, __shfl_xor(mx, ms, 32));
      const float mnew = fmaxf(m_i[v], mx);
      const float corr = __expf(m_i[v] - mnew);
      float rs = 0.f;
#pragma unroll
      for (int nt = 0; nt < 4; ++nt) {
        float p = __expf(sv[nt] - mnew);
        Ps[wv][(v + 8 * half) * 72 + nt * 16 + lx] = f2bf(p);
        rs += p;
      }
#pragma unroll
      for (int ms = 8; ms >= 1; ms >>= 1) rs += __shfl_xor(rs, ms, 32);
      l_i[v] = l_i[v] * corr + rs;
      m_i[v] = mnew;
#pragma unroll
      for (int nt = 0; nt < 8; ++nt) o[nt][v] *= corr;
    }

    // O += P @ V   (P 16x64 as A fragments from wave-private LDS; in-order per wave)
#pragma unroll
    for (int ks = 0; ks < 2; ++ks) {
      const bf16* pb = &Ps[wv][lx * 72 + 32 * ks + half * 16];
      v16bf pa = ld_frag16(pb, pb + 8);
#pragma unroll
      for (int g2 = 0; g2 < 2; ++g2) {
        v16bf bv[4];
#pragma unroll
        for (int j = 0; j < 4; ++j) {
          const int nt = g2 * 4 + j;
          // B[k][n] = V[key = 32*ks + 16*half + e][dv = nt*16+lx]  (contiguous via Vst)
          const bf16* vb = &Vst[(nt * 16 + lx) * 72 + 32 * ks + 16 * half];
          bv[j] = ld_frag16(vb, vb + 8);
        }
#pragma unroll
        for (int j = 0; j < 4; ++j)
          o[g2 * 4 + j] = __builtin_amdgcn_wmma_f32_16x16x32_bf16(false, pa, false, bv[j],
                                                                  (short)0, o[g2 * 4 + j],
                                                                  false, false);
      }
    }
  }

#pragma unroll
  for (int v = 0; v < 8; ++v) {
    const float inv = 1.0f / l_i[v];
    const int grow = qt * 64 + wv * 16 + v + half * 8;
    bf16* orow = Out + (size_t)(b * S_ + grow) * (H_ * DV_) + h * DV_;
#pragma unroll
    for (int nt = 0; nt < 8; ++nt)
      orow[nt * 16 + lx] = f2bf(o[nt][v] * inv);
  }
}

// ---------------------------------------------------------------- orchestration
extern "C" void kernel_launch(void* const* d_in, const int* in_sizes, int n_in,
                              void* d_out, int out_size, void* d_ws, size_t ws_size,
                              hipStream_t stream) {
  (void)in_sizes; (void)n_in; (void)out_size; (void)ws_size;
  const float* hs    = (const float*)d_in[0];
  const float* w_qa  = (const float*)d_in[2];
  const float* b_qa  = (const float*)d_in[3];
  const float* g_q   = (const float*)d_in[4];
  const float* w_qb  = (const float*)d_in[5];
  const float* b_qb  = (const float*)d_in[6];
  const float* w_kva = (const float*)d_in[7];
  const float* b_kva = (const float*)d_in[8];
  const float* g_kv  = (const float*)d_in[9];
  const float* w_kvb = (const float*)d_in[10];
  const float* b_kvb = (const float*)d_in[11];
  const float* w_o   = (const float*)d_in[12];
  const float* b_o   = (const float*)d_in[13];
  float* out = (float*)d_out;

  const int M = B_ * S_;   // 4096
  size_t off = 0;
  auto alloc = [&](size_t elems) -> bf16* {
    bf16* p = (bf16*)((char*)d_ws + off);
    off += (elems * sizeof(bf16) + 255) & ~(size_t)255;
    return p;
  };
  bf16* hs_b   = alloc((size_t)M * D_);
  bf16* wqa_b  = alloc((size_t)D_ * RQ_);
  bf16* wqb_b  = alloc((size_t)RQ_ * (H_ * DH_));
  bf16* wkva_b = alloc((size_t)D_ * (RKV_ + DR_));
  bf16* wkvb_b = alloc((size_t)RKV_ * (HKV_ * (DN_ + DV_)));
  bf16* wo_b   = alloc((size_t)(H_ * DV_) * D_);
  bf16* qa_b   = alloc((size_t)M * RQ_);
  bf16* qan_b  = alloc((size_t)M * RQ_);
  bf16* qb_b   = alloc((size_t)M * (H_ * DH_));
  bf16* kva_b  = alloc((size_t)M * (RKV_ + DR_));
  bf16* kvcn_b = alloc((size_t)M * RKV_);
  bf16* kvb_b  = alloc((size_t)M * (HKV_ * (DN_ + DV_)));
  bf16* Qb     = alloc((size_t)B_ * H_ * S_ * DH_);
  bf16* Kb     = alloc((size_t)B_ * HKV_ * S_ * DH_);
  bf16* Vb     = alloc((size_t)B_ * HKV_ * S_ * DV_);
  bf16* attn_b = alloc((size_t)M * (H_ * DV_));

  // 1) fp32 -> bf16 conversions (8 elems/thread-iter)
  cvt_f32_bf16<<<1024, 256, 0, stream>>>(hs,    hs_b,   ((size_t)M * D_) >> 3);
  cvt_f32_bf16<<<512,  256, 0, stream>>>(w_qa,  wqa_b,  ((size_t)D_ * RQ_) >> 3);
  cvt_f32_bf16<<<512,  256, 0, stream>>>(w_qb,  wqb_b,  ((size_t)RQ_ * (H_ * DH_)) >> 3);
  cvt_f32_bf16<<<512,  256, 0, stream>>>(w_kva, wkva_b, ((size_t)D_ * (RKV_ + DR_)) >> 3);
  cvt_f32_bf16<<<512,  256, 0, stream>>>(w_kvb, wkvb_b, ((size_t)RKV_ * (HKV_ * (DN_ + DV_))) >> 3);
  cvt_f32_bf16<<<1024, 256, 0, stream>>>(w_o,   wo_b,   ((size_t)(H_ * DV_) * D_) >> 3);

  // 2) q_a = hs @ w_qa + b_qa ; rmsnorm ; q_b
  gemm_bf16<<<dim3(RQ_ / 128, M / 128), 256, 0, stream>>>(hs_b, wqa_b, b_qa, qa_b, M, RQ_, D_, 0);
  rmsnorm_bf16<<<M, 256, 0, stream>>>(qa_b, RQ_, g_q, qan_b, RQ_);
  gemm_bf16<<<dim3((H_ * DH_) / 128, M / 128), 256, 0, stream>>>(qan_b, wqb_b, b_qb, qb_b,
                                                                 M, H_ * DH_, RQ_, 0);

  // 3) kv_a = hs @ w_kva + b_kva (N=576) ; rmsnorm of first 512 ; kv_b
  gemm_bf16<<<dim3((RKV_ + DR_ + 127) / 128, M / 128), 256, 0, stream>>>(hs_b, wkva_b, b_kva,
                                                                         kva_b, M, RKV_ + DR_, D_, 0);
  rmsnorm_bf16<<<M, 256, 0, stream>>>(kva_b, RKV_ + DR_, g_kv, kvcn_b, RKV_);
  gemm_bf16<<<dim3((HKV_ * (DN_ + DV_)) / 128, M / 128), 256, 0, stream>>>(kvcn_b, wkvb_b, b_kvb,
                                                                           kvb_b, M, HKV_ * (DN_ + DV_),
                                                                           RKV_, 0);

  // 4) RoPE + assemble Q/K/V
  assemble_q_k <<<4096, 256, 0, stream>>>(qb_b, Qb);
  assemble_kv_k<<<4096, 256, 0, stream>>>(kvb_b, kva_b, Kb, Vb);

  // 5) causal flash attention
  mla_attn<<<dim3(S_ / 64, B_ * H_), 128, 0, stream>>>(Qb, Kb, Vb, attn_b);

  // 6) output projection -> fp32 d_out
  gemm_bf16<<<dim3(D_ / 128, M / 128), 256, 0, stream>>>(attn_b, wo_b, b_o, out, M, D_, H_ * DV_, 1);
}
